// SpaMamba_17325898072383
// MI455X (gfx1250) — compile-verified, matrix operand order checked by reading, not run on hardware
//
#include <hip/hip_runtime.h>
#include <hip/hip_bf16.h>
#include <math.h>

// ---------------------------------------------------------------------------
// SpaMamba for MI455X (gfx1250, wave32, WMMA).
//   D_MODEL=128, D_INNER=256, D_STATE=16, DT_RANK=8, D_CONV=4, L=64, SEQ=512/dir
// ---------------------------------------------------------------------------

typedef __attribute__((ext_vector_type(16))) _Float16 v16h;
typedef __attribute__((ext_vector_type(8)))  float    v8f;

#define TILE_K 64
#define LDSTR  72   // f16 row stride (144B): 16B-aligned groups, conflict-free

// A-matrix gather modes (compile-time -> straight-line address math):
//  0: row pass input  : A[m,k] = x[b,k,h,w], m=(b*64+h)*64+w   (x is NCHW, C=K=128)
//  1: col pass input  : A[m,k] = xr[b,h,w,k], m=(b*64+w)*64+h  (xr is BHWC flat)
//  2: plain row-major : A[m,k] = A[m*K+k]
template <int MODE>
__device__ __forceinline__ float load_A(const float* __restrict__ A,
                                        int m, int k, int K) {
  if (MODE == 0) {
    const int b = m >> 12, h = (m >> 6) & 63, w = m & 63;
    return A[(size_t)b * 524288 + (size_t)k * 4096 + h * 64 + w];
  } else if (MODE == 1) {
    const int b = m >> 12, w = (m >> 6) & 63, h = m & 63;
    return A[(size_t)b * 524288 + (size_t)h * 8192 + w * 128 + k];
  }
  return A[(size_t)m * K + k];
}

// Build one 16x32 (f16) fragment from an LDS tile per ISA 7.12.2:
// lane L: K = kk + ((L>=16)?8:0) + e  (e<8)  and  +16 for e>=8.
__device__ __forceinline__ v16h frag_lds(const _Float16 (*S)[LDSTR],
                                         int row, int kk, int fb) {
  v16h f;
  #pragma unroll
  for (int e = 0; e < 8; ++e) {
    f[e]     = S[row][kk + fb + e];
    f[e + 8] = S[row][kk + 16 + fb + e];
  }
  return f;
}

// C[M,N] = A[M,K] @ W[N,K]^T, 64x64 C-tile per 256-thread block (8 waves),
// K-step 64 -> 4 v_wmma per barrier pair.  Wave w: row tile (w&3),
// col tiles 2*(w>>2), 2*(w>>2)+1.
template <int MODE>
__global__ __launch_bounds__(256) void spamamba_gemm_wmma(
    const float* __restrict__ A, const float* __restrict__ W,
    float* __restrict__ C, int M, int N, int K) {
  __shared__ _Float16 sA[64][LDSTR];
  __shared__ _Float16 sB[64][LDSTR];

  const int tid  = threadIdx.x;
  const int lane = tid & 31;
  const int wid  = tid >> 5;
  const int m0   = blockIdx.y * 64;
  const int n0   = blockIdx.x * 64;
  const int rt   = wid & 3;          // row tile 0..3
  const int ct0  = (wid >> 2) * 2;   // first col tile (0 or 2)

  const int lrow = tid >> 2;         // 0..63: tile row staged by this thread
  const int lk0  = (tid & 3) * 16;   // 16 contiguous k per thread

  v8f acc0 = {}; v8f acc1 = {};

  for (int kb = 0; kb < K; kb += TILE_K) {
    // ---- cooperative stage to LDS (f32 -> f16); contiguous 16-runs merge ----
    #pragma unroll
    for (int i = 0; i < 16; ++i) {
      const int k = lk0 + i;
      sA[lrow][k] = (_Float16)load_A<MODE>(A, m0 + lrow, kb + k, K);
      sB[lrow][k] = (_Float16)W[(size_t)(n0 + lrow) * K + kb + k];
    }
    if (kb + TILE_K < K)  // next weight tile -> global_prefetch_b8
      __builtin_prefetch(&W[(size_t)(n0 + lrow) * K + kb + TILE_K], 0, 0);
    __syncthreads();

    const int fb   = ((lane >> 4) & 1) * 8;
    const int mrow = rt  * 16 + (lane & 15);
    const int nr0  = ct0 * 16 + (lane & 15);
    const int nr1  = nr0 + 16;

    const v16h a0  = frag_lds(sA, mrow, 0,  fb);
    const v16h a1  = frag_lds(sA, mrow, 32, fb);
    const v16h b00 = frag_lds(sB, nr0,  0,  fb);
    const v16h b01 = frag_lds(sB, nr0,  32, fb);
    const v16h b10 = frag_lds(sB, nr1,  0,  fb);
    const v16h b11 = frag_lds(sB, nr1,  32, fb);

    acc0 = __builtin_amdgcn_wmma_f32_16x16x32_f16(false, a0, false, b00,
                                                  (short)0, acc0, false, false);
    acc1 = __builtin_amdgcn_wmma_f32_16x16x32_f16(false, a0, false, b10,
                                                  (short)0, acc1, false, false);
    acc0 = __builtin_amdgcn_wmma_f32_16x16x32_f16(false, a1, false, b01,
                                                  (short)0, acc0, false, false);
    acc1 = __builtin_amdgcn_wmma_f32_16x16x32_f16(false, a1, false, b11,
                                                  (short)0, acc1, false, false);
    __syncthreads();
  }

  // ---- store D (VGPR r: M = base + hi*8 + r, N = lane&15) ----
  const int hi = (lane >> 4) & 1;
  const int mg = m0 + rt * 16 + hi * 8;
  const int ng = n0 + ct0 * 16 + (lane & 15);
  #pragma unroll
  for (int r = 0; r < 8; ++r) {
    C[(size_t)(mg + r) * N + ng]      = acc0[r];
    C[(size_t)(mg + r) * N + ng + 16] = acc1[r];
  }
}

// ---------------------------------------------------------------------------
// Per-sequence fused: depthwise conv + SiLU, x-proj (256->40), dt-proj+softplus,
// selective scan (h[16] per lane), +x*D, *silu(z).  One block per sequence.
// Dynamic LDS: s_raw(64x256) reused as s_dt, s_conv(64x256), s_xdbl(64x40).
// ---------------------------------------------------------------------------
__global__ __launch_bounds__(256) void spamamba_scan(
    const float* __restrict__ xz,     // (SEQ*64, 512): [.,0:256)=xc [.,256:512)=z
    const float* __restrict__ convw,  // (256,1,4)
    const float* __restrict__ convb,  // (256)
    const float* __restrict__ Wx,     // (40,256)
    const float* __restrict__ Wdt,    // (256,8)
    const float* __restrict__ bdt,    // (256)
    const float* __restrict__ Alog,   // (256,16)
    const float* __restrict__ Dp,     // (256)
    float* __restrict__ yz)           // (SEQ*64, 256)
{
  extern __shared__ float smem[];
  float* s_raw  = smem;                 // 64*256 (later reused for dt)
  float* s_conv = smem + 64 * 256;      // 64*256
  float* s_xdbl = smem + 2 * 64 * 256;  // 64*40

  const int tid = threadIdx.x;
  const int seq = blockIdx.x;
  const size_t rb = (size_t)seq * 64;

  // stage xc slice (coalesced: lanes sweep d)
  for (int idx = tid; idx < 64 * 256; idx += 256) {
    const int t = idx >> 8, d = idx & 255;
    s_raw[idx] = xz[(rb + t) * 512 + d];
  }

  // per-channel parameters (thread == channel d)
  const int d = tid;
  const float cw0 = convw[d * 4 + 0], cw1 = convw[d * 4 + 1];
  const float cw2 = convw[d * 4 + 2], cw3 = convw[d * 4 + 3];
  const float cb  = convb[d];
  const float Dd  = Dp[d];
  const float bdtd = bdt[d];
  float wdt[8];
  #pragma unroll
  for (int j = 0; j < 8; ++j) wdt[j] = Wdt[d * 8 + j];
  float A_s[16];
  #pragma unroll
  for (int s = 0; s < 16; ++s) A_s[s] = -__expf(Alog[d * 16 + s]);
  __syncthreads();

  // causal depthwise conv (K=4) + SiLU
  for (int t = 0; t < 64; ++t) {
    float a = cb + cw3 * s_raw[t * 256 + d];
    if (t >= 1) a += cw2 * s_raw[(t - 1) * 256 + d];
    if (t >= 2) a += cw1 * s_raw[(t - 2) * 256 + d];
    if (t >= 3) a += cw0 * s_raw[(t - 3) * 256 + d];
    s_conv[t * 256 + d] = a / (1.f + __expf(-a));
  }
  __syncthreads();

  // xdbl = conv @ Wx^T  (64x40, small -> VALU dot)
  for (int idx = tid; idx < 64 * 40; idx += 256) {
    const int t = idx / 40, j = idx % 40;
    float acc = 0.f;
    for (int dd = 0; dd < 256; ++dd)
      acc += s_conv[t * 256 + dd] * Wx[j * 256 + dd];
    s_xdbl[idx] = acc;
  }
  __syncthreads();

  // dt = softplus(xdbl[:, :8] @ Wdt^T + bdt)  -> reuse s_raw
  for (int t = 0; t < 64; ++t) {
    float p = bdtd;
    #pragma unroll
    for (int j = 0; j < 8; ++j) p += s_xdbl[t * 40 + j] * wdt[j];
    s_raw[t * 256 + d] = (p > 20.f) ? p : __logf(1.f + __expf(p));
  }
  __syncthreads();

  // sequential selective scan; h[16] lives in VGPRs
  float h[16];
  #pragma unroll
  for (int s = 0; s < 16; ++s) h[s] = 0.f;

  for (int t = 0; t < 64; ++t) {
    const float dtv = s_raw[t * 256 + d];
    const float xv  = s_conv[t * 256 + d];
    const float dx  = dtv * xv;
    float y = 0.f;
    #pragma unroll
    for (int s = 0; s < 16; ++s) {
      h[s] = h[s] * __expf(dtv * A_s[s]) + dx * s_xdbl[t * 40 + 8 + s];
      y   += h[s] * s_xdbl[t * 40 + 24 + s];
    }
    y += xv * Dd;
    const float zv = xz[(rb + t) * 512 + 256 + d];
    const float sz = zv / (1.f + __expf(-zv));
    yz[(rb + t) * 256 + d] = y * sz;
  }
}

// ---------------------------------------------------------------------------
// GroupNorm(4 groups) + SiLU; input is col-pass output in (B,W,H,C) flat,
// output is NCHW.  One block per (batch, group).
// ---------------------------------------------------------------------------
__global__ __launch_bounds__(256) void spamamba_gnorm_silu(
    const float* __restrict__ in, const float* __restrict__ gamma,
    const float* __restrict__ beta, float* __restrict__ out) {
  __shared__ float rsum[256];
  __shared__ float rsq[256];
  const int tid = threadIdx.x;
  const int b = blockIdx.x >> 2, g = blockIdx.x & 3;

  float sum = 0.f, sq = 0.f;
  for (int idx = tid; idx < 32 * 64 * 64; idx += 256) {   // c fastest: coalesced read
    const int cl = idx & 31, h = (idx >> 5) & 63, w = idx >> 11;
    const float v = in[(((size_t)b * 64 + w) * 64 + h) * 128 + g * 32 + cl];
    sum += v; sq += v * v;
  }
  rsum[tid] = sum; rsq[tid] = sq;
  __syncthreads();
  for (int s = 128; s > 0; s >>= 1) {
    if (tid < s) { rsum[tid] += rsum[tid + s]; rsq[tid] += rsq[tid + s]; }
    __syncthreads();
  }
  const float inv = 1.f / 131072.f;
  const float mu  = rsum[0] * inv;
  const float var = rsq[0] * inv - mu * mu;
  const float rv  = rsqrtf(var + 1e-5f);

  for (int idx = tid; idx < 32 * 64 * 64; idx += 256) {   // w fastest: coalesced write
    const int w = idx & 63, h = (idx >> 6) & 63, cl = idx >> 12;
    const int c = g * 32 + cl;
    const float v = in[(((size_t)b * 64 + w) * 64 + h) * 128 + c];
    const float y = (v - mu) * rv * gamma[c] + beta[c];
    out[(((size_t)b * 128 + c) * 64 + h) * 64 + w] = y / (1.f + __expf(-y));
  }
}

// ---------------------------------------------------------------------------
extern "C" void kernel_launch(void* const* d_in, const int* in_sizes, int n_in,
                              void* d_out, int out_size, void* d_ws, size_t ws_size,
                              hipStream_t stream) {
  (void)in_sizes; (void)n_in; (void)out_size; (void)ws_size;
  const float* x = (const float*)d_in[0];
  // row params: 1..9, col params: 10..18 (Win,convw,convb,Wx,Wdt,bdt,Alog,D,Wout)
  const float* rWin  = (const float*)d_in[1];
  const float* rCw   = (const float*)d_in[2];
  const float* rCb   = (const float*)d_in[3];
  const float* rWx   = (const float*)d_in[4];
  const float* rWdt  = (const float*)d_in[5];
  const float* rBdt  = (const float*)d_in[6];
  const float* rAlog = (const float*)d_in[7];
  const float* rD    = (const float*)d_in[8];
  const float* rWout = (const float*)d_in[9];
  const float* cWin  = (const float*)d_in[10];
  const float* cCw   = (const float*)d_in[11];
  const float* cCb   = (const float*)d_in[12];
  const float* cWx   = (const float*)d_in[13];
  const float* cWdt  = (const float*)d_in[14];
  const float* cBdt  = (const float*)d_in[15];
  const float* cAlog = (const float*)d_in[16];
  const float* cD    = (const float*)d_in[17];
  const float* cWout = (const float*)d_in[18];
  const float* gnG   = (const float*)d_in[19];
  const float* gnB   = (const float*)d_in[20];
  float* outp = (float*)d_out;

  // workspace carve: xz 64MB | yz 32MB | xr 16MB | colout 16MB
  char* ws = (char*)d_ws;
  float* xz = (float*)(ws);
  float* yz = (float*)(ws + 67108864);
  float* xr = (float*)(ws + 67108864 + 33554432);
  float* co = (float*)(ws + 67108864 + 33554432 + 16777216);

  const int M = 32768;                       // 512 sequences * L=64
  const size_t scan_lds = (2 * 64 * 256 + 64 * 40) * sizeof(float);

  // ---- row pass ----
  spamamba_gemm_wmma<0><<<dim3(8, 512), 256, 0, stream>>>(x,  rWin,  xz, M, 512, 128);
  spamamba_scan<<<512, 256, scan_lds, stream>>>(xz, rCw, rCb, rWx, rWdt, rBdt,
                                                rAlog, rD, yz);
  spamamba_gemm_wmma<2><<<dim3(2, 512), 256, 0, stream>>>(yz, rWout, xr, M, 128, 256);

  // ---- column pass (xr row-major IS (B,H,W,C) flat; mode-1 gather transposes) ----
  spamamba_gemm_wmma<1><<<dim3(8, 512), 256, 0, stream>>>(xr, cWin,  xz, M, 512, 128);
  spamamba_scan<<<512, 256, scan_lds, stream>>>(xz, cCw, cCb, cWx, cWdt, cBdt,
                                                cAlog, cD, yz);
  spamamba_gemm_wmma<2><<<dim3(2, 512), 256, 0, stream>>>(yz, cWout, co, M, 128, 256);

  // ---- group-norm + silu -> NCHW output ----
  spamamba_gnorm_silu<<<32, 256, 0, stream>>>(co, gnG, gnB, outp);
}
